// CausalSelfAttention_72653666779352
// MI455X (gfx1250) — matrix operand order repaired and checked by reading, not compiled, not run on hardware
//
#include <hip/hip_runtime.h>

// ---------------------------------------------------------------------------
// Types for CDNA5 WMMA (gfx1250, wave32)
// ---------------------------------------------------------------------------
typedef __attribute__((ext_vector_type(16))) __bf16 v16bf;
typedef __attribute__((ext_vector_type(8)))  float  v8f;

union Frag16 {
  v16bf    v;
  __bf16   e[16];
  unsigned u[8];
  uint4    q2[2];
};

#define B_   2
#define T_   2048
#define C_   1024
#define NH_  16
#define NKV_ 8
#define HD_  64
#define BT_  (B_ * T_)   // 4096

// CDNA5 async global->LDS DMA (ASYNCcnt). The clang builtin for the load takes
// cuda_device/cuda_shared language-AS pointers that HIP source cannot spell, so
// we issue the instruction with inline asm (ISA 15.18: vdst = LDS byte addr,
// vaddr = 64-bit global addr, "off" = no SADDR). The low 32 bits of a generic
// pointer to a __shared__ object are the LDS byte offset (flat aperture rule).
#if defined(__HIP_DEVICE_COMPILE__) && defined(__gfx1250__) && \
    __has_builtin(__builtin_amdgcn_s_wait_asynccnt)
#define HAVE_ASYNC_LDS 1
#define ASYNC_B128(gp, lp)                                               \
  asm volatile("global_load_async_to_lds_b128 %0, %1, off" ::            \
               "v"((unsigned)(unsigned long long)(lp)),                  \
               "v"((const void*)(gp)) : "memory")
#define ASYNC_WAIT() __builtin_amdgcn_s_wait_asynccnt(0)
#else
#define HAVE_ASYNC_LDS 0
#endif

// ---------------------------------------------------------------------------
// fp32 -> bf16 copy
// ---------------------------------------------------------------------------
__global__ void cvt_f32_bf16(const float* __restrict__ src,
                             __bf16* __restrict__ dst, int n) {
  int i = blockIdx.x * blockDim.x + threadIdx.x;
  if (i < n) dst[i] = (__bf16)src[i];
}

// Pack [Wq | Wk | Wv] -> bf16 [K=1024][N=2048]
__global__ void pack_wqkv(const float* __restrict__ Wq,
                          const float* __restrict__ Wk,
                          const float* __restrict__ Wv,
                          __bf16* __restrict__ dst) {
  int i = blockIdx.x * blockDim.x + threadIdx.x;
  if (i >= 1024 * 2048) return;
  int k = i >> 11;
  int n = i & 2047;
  float val;
  if (n < 1024)      val = Wq[(size_t)k * 1024 + n];
  else if (n < 1536) val = Wk[(size_t)k * 512 + (n - 1024)];
  else               val = Wv[(size_t)k * 512 + (n - 1536)];
  dst[i] = (__bf16)val;
}

// ---------------------------------------------------------------------------
// bf16 GEMM with fp32 output: C[M,N] = A[M,K] x B[K,N]
// 256 threads (8 waves), tile 128(M) x 64(N), K-step 32.
// Wave tile 32x32 = 2x2 v_wmma_f32_16x16x32_bf16 accumulators.
// ---------------------------------------------------------------------------
__global__ __launch_bounds__(256) void gemm_bf16f32(
    const __bf16* __restrict__ A, const __bf16* __restrict__ Bm,
    float* __restrict__ C, int M, int N, int K) {
  __shared__ __bf16 As[128][48];   // [m][k], row stride 96B (16B aligned)
  __shared__ __bf16 Bt[64][40];    // [n][k], transposed, row stride 80B

  const int tid  = threadIdx.x;
  const int lane = tid & 31;
  const int wave = tid >> 5;
  const int wm   = wave & 3;
  const int wn   = wave >> 2;
  const int m0   = blockIdx.y * 128;
  const int n0   = blockIdx.x * 64;
  const int g    = lane >> 4;
  const int ln   = lane & 15;

  v8f acc[2][2];
#pragma unroll
  for (int mt = 0; mt < 2; ++mt)
#pragma unroll
    for (int nt = 0; nt < 2; ++nt)
#pragma unroll
      for (int r = 0; r < 8; ++r) acc[mt][nt][r] = 0.f;

  // Staging coordinates (A: 128 rows x 4 uint4; B: 32 rows x 8 uint4)
  const int arow0 = (tid) >> 2,        aq40 = (tid) & 3;
  const int arow1 = (tid + 256) >> 2,  aq41 = (tid + 256) & 3;
  const int bkr   = tid >> 3,          bq4  = tid & 7;

  for (int k0 = 0; k0 < K; k0 += 32) {
    __syncthreads();

#if HAVE_ASYNC_LDS
    // Async DMA global -> LDS (b128), tracked by ASYNCcnt
    ASYNC_B128(A + (size_t)(m0 + arow0) * K + k0 + aq40 * 8,
               &As[arow0][aq40 * 8]);
    ASYNC_B128(A + (size_t)(m0 + arow1) * K + k0 + aq41 * 8,
               &As[arow1][aq41 * 8]);
#else
    uint4 ta0 = *(const uint4*)(A + (size_t)(m0 + arow0) * K + k0 + aq40 * 8);
    uint4 ta1 = *(const uint4*)(A + (size_t)(m0 + arow1) * K + k0 + aq41 * 8);
#endif
    // B tile: load wide, transpose through VGPRs into Bt[n][k]
    union { uint4 v; __bf16 e[8]; } ub;
    ub.v = *(const uint4*)(Bm + (size_t)(k0 + bkr) * N + n0 + bq4 * 8);
#if !HAVE_ASYNC_LDS
    *(uint4*)&As[arow0][aq40 * 8] = ta0;
    *(uint4*)&As[arow1][aq41 * 8] = ta1;
#endif
#pragma unroll
    for (int j = 0; j < 8; ++j) Bt[bq4 * 8 + j][bkr] = ub.e[j];

#if HAVE_ASYNC_LDS
    ASYNC_WAIT();
#endif
    __syncthreads();

    // Fragments (b128 LDS reads; layouts per CDNA5 ISA 7.12.2)
    Frag16 af[2], bfr[2];
#pragma unroll
    for (int mt = 0; mt < 2; ++mt) {
      int row = wm * 32 + mt * 16 + ln;
#pragma unroll
      for (int half = 0; half < 2; ++half)
        af[mt].q2[half] = *(const uint4*)&As[row][half * 16 + g * 8];
    }
#pragma unroll
    for (int nt = 0; nt < 2; ++nt) {
      int col = wn * 32 + nt * 16 + ln;
#pragma unroll
      for (int j2 = 0; j2 < 2; ++j2)
        bfr[nt].q2[j2] = *(const uint4*)&Bt[col][j2 * 16 + g * 8];
    }
#pragma unroll
    for (int mt = 0; mt < 2; ++mt)
#pragma unroll
      for (int nt = 0; nt < 2; ++nt)
        acc[mt][nt] = __builtin_amdgcn_wmma_f32_16x16x32_bf16(
            false, af[mt].v, false, bfr[nt].v, (short)0, acc[mt][nt],
            false, false);
  }

#pragma unroll
  for (int mt = 0; mt < 2; ++mt)
#pragma unroll
    for (int nt = 0; nt < 2; ++nt)
#pragma unroll
      for (int r = 0; r < 8; ++r) {
        int row = m0 + wm * 32 + mt * 16 + r + 8 * g;
        int col = n0 + wn * 32 + nt * 16 + ln;
        C[(size_t)row * N + col] = acc[mt][nt][r];
      }
}

// ---------------------------------------------------------------------------
// Epilogue: gate/ve add, RoPE, RMSNorm; emit bf16 q/k/v.
// ---------------------------------------------------------------------------
__device__ static inline void rope_norm(const float* __restrict__ src,
                                        const float* __restrict__ cs,
                                        const float* __restrict__ sn,
                                        __bf16* __restrict__ dst) {
  float v[64];
  float ss = 0.f;
#pragma unroll
  for (int d = 0; d < 32; ++d) {
    float x1 = src[d], x2 = src[d + 32];
    float c = cs[d], s = sn[d];
    float o1 =  x1 * c + x2 * s;
    float o2 = -x1 * s + x2 * c;
    v[d] = o1; v[d + 32] = o2;
    ss += o1 * o1 + o2 * o2;
  }
  float r = rsqrtf(ss * (1.f / 64.f) + 1.1920929e-7f);
#pragma unroll
  for (int d = 0; d < 64; ++d) dst[d] = (__bf16)(v[d] * r);
}

__global__ __launch_bounds__(256) void qkv_epilogue(
    const float* __restrict__ qkv,   // [4096][2048]
    const float* __restrict__ x,     // [4096][1024]
    const float* __restrict__ ve,    // [4096][512]
    const float* __restrict__ cosb,  // [2048][32]
    const float* __restrict__ sinb,  // [2048][32]
    const float* __restrict__ Wgate, // [32][8]
    __bf16* __restrict__ qb,         // [4096][1024]
    __bf16* __restrict__ kb,         // [4096][512]
    __bf16* __restrict__ vb) {       // [4096][512]
  int gid   = blockIdx.x * blockDim.x + threadIdx.x;  // token*32 + unit
  int token = gid >> 5;
  int unit  = gid & 31;
  int t     = token & (T_ - 1);
  const float* row = qkv + (size_t)token * 2048;
  const float* cs = cosb + (size_t)t * 32;
  const float* sn = sinb + (size_t)t * 32;

  if (unit < 16) {
    rope_norm(row + unit * 64, cs, sn, qb + (size_t)token * 1024 + unit * 64);
  } else if (unit < 24) {
    int hk = unit - 16;
    rope_norm(row + 1024 + hk * 64, cs, sn, kb + (size_t)token * 512 + hk * 64);
  } else {
    int hk = unit - 24;
    float dot = 0.f;
    const float* xr = x + (size_t)token * 1024;
#pragma unroll
    for (int i = 0; i < 32; ++i) dot += xr[i] * Wgate[i * 8 + hk];
    float gate = 2.f / (1.f + __expf(-dot));
    const float* vep = ve + (size_t)token * 512 + hk * 64;
    __bf16* vo = vb + (size_t)token * 512 + hk * 64;
#pragma unroll
    for (int d = 0; d < 64; ++d)
      vo[d] = (__bf16)(row[1536 + hk * 64 + d] + gate * vep[d]);
  }
}

// ---------------------------------------------------------------------------
// Flash attention: block = 4 waves (128 thr), each wave owns 16 q rows.
// Grid: (T/64, B*NH). Heads 0-7 causal (global); 8-15 causal + window.
// ---------------------------------------------------------------------------
__global__ __launch_bounds__(128) void attn_kernel(
    const __bf16* __restrict__ qb,   // [4096][1024]
    const __bf16* __restrict__ kb,   // [4096][512]
    const __bf16* __restrict__ vb,   // [4096][512]
    __bf16* __restrict__ yb,         // [4096][1024]
    const int* __restrict__ winp) {
  const int tid  = threadIdx.x;
  const int lane = tid & 31;
  const int wave = tid >> 5;
  const int qblk = blockIdx.x;
  const int bh   = blockIdx.y;
  const int b    = bh >> 4;
  const int h    = bh & 15;
  const int hk   = h >> 1;
  const bool is_local = (h >= 8);
  const int window = *winp;

  const int row0 = qblk * 64;
  const int qt   = row0 + wave * 16;
  const int g    = lane >> 4;
  const int ln   = lane & 15;

  __shared__ __bf16 Ks[32][80];       // [key][d], row stride 160B
  __shared__ __bf16 Vt[64][40];       // [d][key], row stride 80B
  __shared__ __bf16 Pl[4][16][40];    // per-wave P staging, row stride 80B

  // Q fragments: per-lane data is memory-contiguous -> two b128 per K-step
  Frag16 aq[2];
  {
    const __bf16* qp = qb + (size_t)(b * T_ + qt + ln) * 1024 + h * 64;
#pragma unroll
    for (int s = 0; s < 2; ++s)
#pragma unroll
      for (int half = 0; half < 2; ++half)
        aq[s].q2[half] = *(const uint4*)(qp + s * 32 + half * 16 + g * 8);
  }

  v8f acco[4], m_i, l_i;
#pragma unroll
  for (int tt = 0; tt < 4; ++tt)
#pragma unroll
    for (int r = 0; r < 8; ++r) acco[tt][r] = 0.f;
#pragma unroll
  for (int r = 0; r < 8; ++r) { m_i[r] = -1e30f; l_i[r] = 0.f; }

  int lo = 0;
  if (is_local) { int w0 = row0 - window; if (w0 > 0) lo = w0 & ~31; }
  const int hi = row0 + 64;

  // Staging coordinates: 32 keys x 8 uint4 = 256 positions / 128 threads
  const int key0 = (tid) >> 3,       q40 = (tid) & 7;
  const int key1 = (tid + 128) >> 3, q41 = (tid + 128) & 7;

  for (int c0 = lo; c0 < hi; c0 += 32) {
    __syncthreads();
    {
      size_t base0 = (size_t)(b * T_ + c0 + key0) * 512 + hk * 64 + q40 * 8;
      size_t base1 = (size_t)(b * T_ + c0 + key1) * 512 + hk * 64 + q41 * 8;
#if HAVE_ASYNC_LDS
      ASYNC_B128(kb + base0, &Ks[key0][q40 * 8]);
      ASYNC_B128(kb + base1, &Ks[key1][q41 * 8]);
#else
      uint4 tk0 = *(const uint4*)(kb + base0);
      uint4 tk1 = *(const uint4*)(kb + base1);
#endif
      // V: wide loads, transpose through VGPRs
      union { uint4 v; __bf16 e[8]; } tv0, tv1;
      tv0.v = *(const uint4*)(vb + base0);
      tv1.v = *(const uint4*)(vb + base1);
#if !HAVE_ASYNC_LDS
      *(uint4*)&Ks[key0][q40 * 8] = tk0;
      *(uint4*)&Ks[key1][q41 * 8] = tk1;
#endif
#pragma unroll
      for (int j = 0; j < 8; ++j) Vt[q40 * 8 + j][key0] = tv0.e[j];
#pragma unroll
      for (int j = 0; j < 8; ++j) Vt[q41 * 8 + j][key1] = tv1.e[j];
#if HAVE_ASYNC_LDS
      ASYNC_WAIT();
#endif
    }
    __syncthreads();

    bool active = (c0 <= qt + 15);
    if (is_local) active = active && (c0 + 31 >= qt - window);
    if (active) {
      // S = Q K^T : 2 key-tiles x 2 dsteps
      v8f sacc[2];
#pragma unroll
      for (int nt = 0; nt < 2; ++nt) {
#pragma unroll
        for (int r = 0; r < 8; ++r) sacc[nt][r] = 0.f;
        int key = nt * 16 + ln;
#pragma unroll
        for (int s = 0; s < 2; ++s) {
          Frag16 bk;
          int dbase = s * 32 + g * 8;
#pragma unroll
          for (int j2 = 0; j2 < 2; ++j2)
            bk.q2[j2] = *(const uint4*)&Ks[key][dbase + 16 * j2];
          sacc[nt] = __builtin_amdgcn_wmma_f32_16x16x32_bf16(
              false, aq[s].v, false, bk.v, (short)0, sacc[nt], false, false);
        }
      }

      // Scale + mask; masked entries get p=0 (robust vs fully-masked rows)
      float sm[2][8]; bool msk[2][8];
#pragma unroll
      for (int nt = 0; nt < 2; ++nt) {
        int j = c0 + nt * 16 + ln;
#pragma unroll
        for (int r = 0; r < 8; ++r) {
          int i_row = qt + r + 8 * g;
          float sv = sacc[nt][r] * 0.125f;  // 1/sqrt(64)
          bool ok = (j <= i_row);
          if (is_local) ok = ok && (j >= i_row - window);
          msk[nt][r] = ok;
          sm[nt][r]  = ok ? sv : -1e30f;
        }
      }

      // Online softmax: reductions across the 16-lane halves (wave32)
      v8f mnew, scalef;
#pragma unroll
      for (int r = 0; r < 8; ++r) {
        float rm = fmaxf(sm[0][r], sm[1][r]);
#pragma unroll
        for (int off = 1; off < 16; off <<= 1)
          rm = fmaxf(rm, __shfl_xor(rm, off, 32));
        mnew[r]   = fmaxf(m_i[r], rm);
        scalef[r] = __expf(m_i[r] - mnew[r]);
      }
      m_i = mnew;

      float rowsum[8];
#pragma unroll
      for (int r = 0; r < 8; ++r) rowsum[r] = 0.f;
#pragma unroll
      for (int nt = 0; nt < 2; ++nt)
#pragma unroll
        for (int r = 0; r < 8; ++r) {
          float p = msk[nt][r] ? __expf(sm[nt][r] - mnew[r]) : 0.f;
          Pl[wave][r + 8 * g][nt * 16 + ln] = (__bf16)p;
          rowsum[r] += p;
        }
#pragma unroll
      for (int r = 0; r < 8; ++r) {
        float rs = rowsum[r];
#pragma unroll
        for (int off = 1; off < 16; off <<= 1)
          rs += __shfl_xor(rs, off, 32);
        l_i[r] = l_i[r] * scalef[r] + rs;
      }
#pragma unroll
      for (int tt = 0; tt < 4; ++tt)
#pragma unroll
        for (int r = 0; r < 8; ++r) acco[tt][r] *= scalef[r];

      // Per-wave LDS round trip for P: explicit CDNA5 split DS counter wait
      asm volatile("s_wait_dscnt 0" ::: "memory");

      Frag16 ap;
#pragma unroll
      for (int half = 0; half < 2; ++half)
        ap.q2[half] = *(const uint4*)&Pl[wave][ln][half * 16 + g * 8];

      // O += P V : 4 d-tiles, K=32 keys
#pragma unroll
      for (int tt = 0; tt < 4; ++tt) {
        Frag16 bv;
        int dcol = tt * 16 + ln;
#pragma unroll
        for (int j2 = 0; j2 < 2; ++j2)
          bv.q2[j2] = *(const uint4*)&Vt[dcol][j2 * 16 + g * 8];
        acco[tt] = __builtin_amdgcn_wmma_f32_16x16x32_bf16(
            false, ap.v, false, bv.v, (short)0, acco[tt], false, false);
      }
    }
  }

  // Normalize and store bf16 y
#pragma unroll
  for (int tt = 0; tt < 4; ++tt)
#pragma unroll
    for (int r = 0; r < 8; ++r) {
      int row = qt + r + 8 * g;
      float o = acco[tt][r] / l_i[r];
      yb[(size_t)(b * T_ + row) * 1024 + h * 64 + tt * 16 + ln] = (__bf16)o;
    }
}

// ---------------------------------------------------------------------------
// Launch
// ---------------------------------------------------------------------------
extern "C" void kernel_launch(void* const* d_in, const int* in_sizes, int n_in,
                              void* d_out, int out_size, void* d_ws,
                              size_t ws_size, hipStream_t stream) {
  const float* x     = (const float*)d_in[0];
  const float* ve    = (const float*)d_in[1];
  const float* cosb  = (const float*)d_in[2];
  const float* sinb  = (const float*)d_in[3];
  const float* Wq    = (const float*)d_in[4];
  const float* Wk    = (const float*)d_in[5];
  const float* Wv    = (const float*)d_in[6];
  const float* Wproj = (const float*)d_in[7];
  const float* Wgate = (const float*)d_in[8];
  const int*   winp  = (const int*)d_in[9];
  float* out = (float*)d_out;

  if (ws_size < (size_t)73400320) return;
  char* w = (char*)d_ws;
  __bf16* xb     = (__bf16*)(w);              // 4096x1024 bf16 =  8 MB
  __bf16* wqkv   = (__bf16*)(w + 8388608);    // 1024x2048 bf16 =  4 MB
  __bf16* wprojb = (__bf16*)(w + 12582912);   // 1024x1024 bf16 =  2 MB
  float*  qkvf   = (float* )(w + 14680064);   // 4096x2048 f32  = 32 MB
  __bf16* qb     = (__bf16*)(w + 48234496);   // 4096x1024 bf16 =  8 MB
  __bf16* kb     = (__bf16*)(w + 56623104);   // 4096x512  bf16 =  4 MB
  __bf16* vb     = (__bf16*)(w + 60817408);   // 4096x512  bf16 =  4 MB
  __bf16* yb     = (__bf16*)(w + 65011712);   // 4096x1024 bf16 =  8 MB

  cvt_f32_bf16<<<(4194304 + 255) / 256, 256, 0, stream>>>(x, xb, 4194304);
  pack_wqkv<<<(2097152 + 255) / 256, 256, 0, stream>>>(Wq, Wk, Wv, wqkv);
  cvt_f32_bf16<<<(1048576 + 255) / 256, 256, 0, stream>>>(Wproj, wprojb,
                                                          1048576);
  gemm_bf16f32<<<dim3(2048 / 64, 4096 / 128), 256, 0, stream>>>(
      xb, wqkv, qkvf, 4096, 2048, 1024);
  qkv_epilogue<<<512, 256, 0, stream>>>(qkvf, x, ve, cosb, sinb, Wgate, qb, kb,
                                        vb);
  attn_kernel<<<dim3(T_ / 64, B_ * NH_), 128, 0, stream>>>(qb, kb, vb, yb,
                                                           winp);
  gemm_bf16f32<<<dim3(1024 / 64, 4096 / 128), 256, 0, stream>>>(
      yb, wprojb, out, 4096, 1024, 1024);
}
